// ChatGLM2Attention_77695958385103
// MI455X (gfx1250) — matrix operand order repaired and checked by reading, not compiled, not run on hardware
//
#include <hip/hip_runtime.h>
#include <hip/hip_fp16.h>

typedef __attribute__((ext_vector_type(16))) _Float16 v16h;
typedef __attribute__((ext_vector_type(8)))  _Float16 v8h;
typedef __attribute__((ext_vector_type(4)))  _Float16 v4h;
typedef __attribute__((ext_vector_type(8)))  float    v8f;
typedef __attribute__((ext_vector_type(4)))  int      v4i_;

#define B_   2
#define S_   1024
#define H_   4096
#define NH_  32
#define G_   2
#define HD_  128
#define QKVW (NH_*HD_ + 2*G_*HD_)   // 4608

union FragA { v16h v; v8h h[2]; };

// ---- CDNA5 async global->LDS copy (probe via __has_builtin; fall back) ----
#if defined(__has_builtin)
#if __has_builtin(__builtin_amdgcn_global_load_async_to_lds_b128)
#define HAVE_ASYNC_LDS 1
#endif
#if __has_builtin(__builtin_amdgcn_s_wait_asynccnt)
#define HAVE_WAIT_ASYNC 1
#endif
#endif

#ifdef HAVE_ASYNC_LDS
typedef __attribute__((address_space(1))) v4i_* glb_v4i_p;   // global (AS1)
typedef __attribute__((address_space(3))) v4i_* lds_v4i_p;   // LDS (AS3)
__device__ __forceinline__ void async_copy16(const _Float16* g, _Float16* l) {
    __builtin_amdgcn_global_load_async_to_lds_b128(
        (glb_v4i_p)(g), (lds_v4i_p)(l), 0, 0);
}
__device__ __forceinline__ void async_wait0() {
#ifdef HAVE_WAIT_ASYNC
    __builtin_amdgcn_s_wait_asynccnt(0);
#else
    asm volatile("s_wait_asynccnt 0x0" ::: "memory");
#endif
}
#endif

// ---------------------------------------------------------------------------
// fp32 -> fp16 conversion (vectorized x4)
// ---------------------------------------------------------------------------
__global__ void cvt_f32_f16(const float* __restrict__ in, _Float16* __restrict__ out, int n4) {
    int i = blockIdx.x * blockDim.x + threadIdx.x;
    if (i < n4) {
        float4 f = reinterpret_cast<const float4*>(in)[i];
        v4h h;
        h.x = (_Float16)f.x; h.y = (_Float16)f.y;
        h.z = (_Float16)f.z; h.w = (_Float16)f.w;
        reinterpret_cast<v4h*>(out)[i] = h;
    }
}

// ---------------------------------------------------------------------------
// Tiled WMMA GEMM: C[M,N] (fp32) = A[M,K](f16) * B[K,N](f16) (+ bias)
// 128x128 block tile, 256 threads = 8 waves, each wave computes 32x64.
// A tile staged via CDNA5 async global->LDS (ASYNCcnt); B tile staged through
// VGPRs with an in-register transpose to [n][k] so both A and B WMMA
// fragments are contiguous 16B ds_load_b128 chunks per lane.
// ---------------------------------------------------------------------------
__global__ __launch_bounds__(256)
void gemm_f16f32(const _Float16* __restrict__ A, const _Float16* __restrict__ Bm,
                 const float* __restrict__ bias, float* __restrict__ C,
                 int M, int N, int K) {
    __shared__ __align__(32) _Float16 As[128 * 32];
    __shared__ __align__(32) _Float16 Bs[128 * 32];   // [n][k] transposed

    const int tid  = threadIdx.x;
    const int lane = tid & 31;
    const int wave = tid >> 5;
    const int ln   = lane & 15;
    const bool hiL = lane >= 16;
    const int m0 = blockIdx.y * 128;
    const int n0 = blockIdx.x * 128;
    const int wm = (wave >> 1) * 32;   // 0,32,64,96
    const int wn = (wave & 1) * 64;    // 0,64
    // ISA 16-bit A 16x32 layout: lane<16 -> K {0..7,16..23}; lane>=16 -> {8..15,24..31}
    const int aoff = hiL ? 8 : 0;
    // ISA 16-bit B 32x16 layout: lane<16 -> K {0..7,16..23}; lane>=16 -> {8..15,24..31} (on Bt rows)
    const int boff = hiL ? 8 : 0;

    // per-thread staging coordinates (hoisted out of the K loop)
    const int ar = tid >> 2, ac = (tid & 3) * 8;      // A: 2 rows (ar, ar+64)
    const int br = tid >> 4, bc = (tid & 15) * 8;     // B: 2 rows (br, br+16)
    const _Float16* Ag0 = &A[(size_t)(m0 + ar) * K + ac];
    const _Float16* Ag1 = &A[(size_t)(m0 + ar + 64) * K + ac];
    const _Float16* Bg0 = &Bm[(size_t)br * N + n0 + bc];
    const _Float16* Bg1 = &Bm[(size_t)(br + 16) * N + n0 + bc];
    _Float16* AsD0 = &As[ar * 32 + ac];
    _Float16* AsD1 = &As[(ar + 64) * 32 + ac];

    v8f zc = {};
    v8f acc[2][4];
    for (int i = 0; i < 2; ++i)
        for (int j = 0; j < 4; ++j) acc[i][j] = zc;

    for (int kt = 0; kt < K; kt += 32) {
        // ---- stage tiles into LDS ----
#ifdef HAVE_ASYNC_LDS
        async_copy16(Ag0 + kt, AsD0);                 // A tile: async, no VGPR trip
        async_copy16(Ag1 + kt, AsD1);
        v8h b0 = *reinterpret_cast<const v8h*>(Bg0 + (size_t)kt * N);
        v8h b1 = *reinterpret_cast<const v8h*>(Bg1 + (size_t)kt * N);
#else
        v8h a0 = *reinterpret_cast<const v8h*>(Ag0 + kt);
        v8h a1 = *reinterpret_cast<const v8h*>(Ag1 + kt);
        v8h b0 = *reinterpret_cast<const v8h*>(Bg0 + (size_t)kt * N);
        v8h b1 = *reinterpret_cast<const v8h*>(Bg1 + (size_t)kt * N);
        *reinterpret_cast<v8h*>(AsD0) = a0;
        *reinterpret_cast<v8h*>(AsD1) = a1;
#endif
        #pragma unroll
        for (int j = 0; j < 8; ++j) Bs[(bc + j) * 32 + br] = b0[j];
        #pragma unroll
        for (int j = 0; j < 8; ++j) Bs[(bc + j) * 32 + br + 16] = b1[j];

        if (kt + 32 < K) {   // global_prefetch_b8 of next K-slab
            __builtin_prefetch(Ag0 + kt + 32, 0, 3);
            __builtin_prefetch(Bg0 + (size_t)(kt + 32) * N, 0, 3);
        }
#ifdef HAVE_ASYNC_LDS
        async_wait0();
#endif
        __syncthreads();

        FragA a[2];
        #pragma unroll
        for (int mt = 0; mt < 2; ++mt) {
            const _Float16* p = &As[(wm + mt * 16 + ln) * 32];
            a[mt].h[0] = *reinterpret_cast<const v8h*>(p + aoff);
            a[mt].h[1] = *reinterpret_cast<const v8h*>(p + aoff + 16);
        }
        #pragma unroll
        for (int nt = 0; nt < 4; ++nt) {
            FragA b;
            const _Float16* p = &Bs[(wn + nt * 16 + ln) * 32];
            b.h[0] = *reinterpret_cast<const v8h*>(p + boff);
            b.h[1] = *reinterpret_cast<const v8h*>(p + boff + 16);
            #pragma unroll
            for (int mt = 0; mt < 2; ++mt)
                acc[mt][nt] = __builtin_amdgcn_wmma_f32_16x16x32_f16(
                    false, a[mt].v, false, b.v, (short)0, acc[mt][nt], false, false);
        }
        __syncthreads();
    }

    // ---- epilogue: C frag rows: lane<16 -> M=r, lane>=16 -> M=8+r; N=lane&15
    const int mrow = hiL ? 8 : 0;
    for (int mt = 0; mt < 2; ++mt)
        for (int nt = 0; nt < 4; ++nt) {
            int n = n0 + wn + nt * 16 + ln;
            float badd = bias ? bias[n] : 0.f;
            #pragma unroll
            for (int r = 0; r < 8; ++r) {
                int m = m0 + wm + mt * 16 + mrow + r;
                C[(size_t)m * N + n] = acc[mt][nt][r] + badd;
            }
        }
}

// ---------------------------------------------------------------------------
// RoPE + split: qkv fp32 [B,S,4608] -> qh[b,h,s,d], kh[b,g,s,d] (f16, rope on
// first 64 dims), vt[b,g,d,s] (f16 transposed for P@V B-fragments)
// ---------------------------------------------------------------------------
__global__ void rope_split(const float* __restrict__ qkv, const float* __restrict__ rc,
                           _Float16* __restrict__ qh, _Float16* __restrict__ kh,
                           _Float16* __restrict__ vt) {
    int idx = blockIdx.x * blockDim.x + threadIdx.x;
    if (idx >= B_ * S_ * QKVW) return;
    int col = idx % QKVW;
    int bs  = idx / QKVW;
    int s = bs % S_, b = bs / S_;
    float x = qkv[idx];
    int d = col & 127;
    bool isv = col >= NH_ * HD_ + G_ * HD_;           // v columns: 4352..4607
    if (!isv && d < 64) {                             // interleaved RoPE pairs
        float x2 = qkv[idx ^ 1];                      // partner within (d even/odd)
        int i = d >> 1;
        float c0 = rc[((size_t)(s * B_ + b) * 32 + i) * 2 + 0];
        float c1 = rc[((size_t)(s * B_ + b) * 32 + i) * 2 + 1];
        x = (d & 1) ? (x * c0 + x2 * c1) : (x * c0 - x2 * c1);
    }
    if (col < NH_ * HD_) {
        int h = col >> 7;
        qh[(((size_t)b * NH_ + h) * S_ + s) * HD_ + d] = (_Float16)x;
    } else if (!isv) {
        int g = (col - NH_ * HD_) >> 7;
        kh[(((size_t)b * G_ + g) * S_ + s) * HD_ + d] = (_Float16)x;
    } else {
        int g = (col - NH_ * HD_ - G_ * HD_) >> 7;
        vt[(((size_t)b * G_ + g) * HD_ + d) * S_ + s] = (_Float16)x;
    }
}

// ---------------------------------------------------------------------------
// Flash attention: grid (S/64, B*NH), 128 threads = 4 waves.
// Each wave owns 16 query rows; online softmax; WMMA for Q@K^T and P@V.
// kh rows [key][d] directly match the B^t fragment layout for Q@K^T; vt rows
// [d][key] match it for P@V. P is re-shaped into A-frag layout via LDS.
// ---------------------------------------------------------------------------
__global__ __launch_bounds__(128)
void attn_kernel(const _Float16* __restrict__ qh, const _Float16* __restrict__ kh,
                 const _Float16* __restrict__ vt, _Float16* __restrict__ ctx) {
    __shared__ __align__(32) _Float16 Pbuf[4 * 16 * 64];
    const int lane = threadIdx.x & 31;
    const int wave = threadIdx.x >> 5;
    const int ln   = lane & 15;
    const bool hiL = lane >= 16;
    const int qt = blockIdx.x;
    const int bh = blockIdx.y;
    const int b = bh / NH_;
    const int h = bh % NH_;
    const int g = h / (NH_ / G_);
    const int q0 = qt * 64 + wave * 16;
    const float scale = 0.08838834764831845f;   // 1/sqrt(128)
    const int aoff = hiL ? 8 : 0;
    const int boff = hiL ? 8 : 0;
    const int mrow = hiL ? 8 : 0;

    // preload Q A-fragments (16 rows x 128 d = 4 K-chunks of 32)
    FragA aq[4];
    const _Float16* qbase = qh + (((size_t)bh) * S_ + q0 + ln) * HD_;
    #pragma unroll
    for (int dk = 0; dk < 4; ++dk) {
        aq[dk].h[0] = *reinterpret_cast<const v8h*>(qbase + dk * 32 + aoff);
        aq[dk].h[1] = *reinterpret_cast<const v8h*>(qbase + dk * 32 + aoff + 16);
    }

    v8f zc = {};
    v8f O[8];
    for (int i = 0; i < 8; ++i) O[i] = zc;
    float rmax[8], rsum[8];
    for (int r = 0; r < 8; ++r) { rmax[r] = -1e30f; rsum[r] = 0.f; }

    const _Float16* kbase = kh + ((size_t)(b * G_ + g)) * S_ * HD_;
    const _Float16* vbase = vt + ((size_t)(b * G_ + g)) * HD_ * S_;
    _Float16* pw = &Pbuf[wave * 16 * 64];

    for (int kt = 0; kt <= qt; ++kt) {
        const int k0 = kt * 64;
        const bool diag = (kt == qt);

        // ---- S = Q @ K^T : 16x64 tile as 4 C-fragments ----
        v8f sc[4];
        #pragma unroll
        for (int nb = 0; nb < 4; ++nb) {
            v8f c = zc;
            const _Float16* kr = kbase + (size_t)(k0 + nb * 16 + ln) * HD_;
            #pragma unroll
            for (int dk = 0; dk < 4; ++dk) {
                FragA bk;
                bk.h[0] = *reinterpret_cast<const v8h*>(kr + dk * 32 + boff);
                bk.h[1] = *reinterpret_cast<const v8h*>(kr + dk * 32 + boff + 16);
                c = __builtin_amdgcn_wmma_f32_16x16x32_f16(
                        false, aq[dk].v, false, bk.v, (short)0, c, false, false);
            }
            sc[nb] = c;
        }

        // ---- scale + causal mask (diagonal tile only) ----
        #pragma unroll
        for (int nb = 0; nb < 4; ++nb) {
            #pragma unroll
            for (int r = 0; r < 8; ++r) {
                float v = sc[nb][r] * scale;
                if (diag) {
                    int qrow = q0 + mrow + r;
                    int key  = k0 + nb * 16 + ln;
                    if (key > qrow) v = -1e30f;
                }
                sc[nb][r] = v;
            }
        }

        // ---- online softmax per row slot (rows live in 16-lane half-groups) ----
        #pragma unroll
        for (int r = 0; r < 8; ++r) {
            float tmax = fmaxf(fmaxf(sc[0][r], sc[1][r]), fmaxf(sc[2][r], sc[3][r]));
            for (int off = 8; off >= 1; off >>= 1)
                tmax = fmaxf(tmax, __shfl_xor(tmax, off));
            float nm   = fmaxf(rmax[r], tmax);
            float corr = __expf(rmax[r] - nm);
            float psum = 0.f;
            #pragma unroll
            for (int nb = 0; nb < 4; ++nb) {
                float p = __expf(sc[nb][r] - nm);
                psum += p;
                pw[(mrow + r) * 64 + nb * 16 + ln] = (_Float16)p;
            }
            for (int off = 8; off >= 1; off >>= 1)
                psum += __shfl_xor(psum, off);
            rsum[r] = rsum[r] * corr + psum;
            rmax[r] = nm;
            #pragma unroll
            for (int db = 0; db < 8; ++db) O[db][r] *= corr;
        }

        // ---- O += P(16x64) @ V(64x128) ----
        FragA pa[2];
        #pragma unroll
        for (int kk = 0; kk < 2; ++kk) {
            const _Float16* pr = &pw[ln * 64 + kk * 32];
            pa[kk].h[0] = *reinterpret_cast<const v8h*>(pr + aoff);
            pa[kk].h[1] = *reinterpret_cast<const v8h*>(pr + aoff + 16);
        }
        #pragma unroll
        for (int db = 0; db < 8; ++db) {
            const _Float16* vr = vbase + (size_t)(db * 16 + ln) * S_ + k0;
            #pragma unroll
            for (int kk = 0; kk < 2; ++kk) {
                FragA bv;
                bv.h[0] = *reinterpret_cast<const v8h*>(vr + kk * 32 + boff);
                bv.h[1] = *reinterpret_cast<const v8h*>(vr + kk * 32 + boff + 16);
                O[db] = __builtin_amdgcn_wmma_f32_16x16x32_f16(
                            false, pa[kk].v, false, bv.v, (short)0, O[db], false, false);
            }
        }
    }

    // ---- epilogue: ctx[b, q, h*128 + d] f16 ----
    #pragma unroll
    for (int db = 0; db < 8; ++db) {
        int d = db * 16 + ln;
        #pragma unroll
        for (int r = 0; r < 8; ++r) {
            int q = q0 + mrow + r;
            float v = O[db][r] / rsum[r];
            ctx[((size_t)(b * S_ + q)) * H_ + h * HD_ + d] = (_Float16)v;
        }
    }
}

// ---------------------------------------------------------------------------
// Launch
// ---------------------------------------------------------------------------
extern "C" void kernel_launch(void* const* d_in, const int* in_sizes, int n_in,
                              void* d_out, int out_size, void* d_ws, size_t ws_size,
                              hipStream_t stream) {
    (void)in_sizes; (void)n_in; (void)out_size; (void)ws_size;
    const float* hs     = (const float*)d_in[0];
    const float* rope   = (const float*)d_in[1];
    const float* wqkv   = (const float*)d_in[2];
    const float* bqkv   = (const float*)d_in[3];
    const float* wdense = (const float*)d_in[4];
    float* out = (float*)d_out;

    char* ws = (char*)d_ws;
    size_t off = 0;
    auto alloc = [&](size_t bytes) -> char* {
        char* p = ws + off;
        off += (bytes + 255) & ~(size_t)255;
        return p;
    };
    _Float16* hs16   = (_Float16*)alloc((size_t)B_ * S_ * H_ * 2);
    _Float16* wqkv16 = (_Float16*)alloc((size_t)H_ * QKVW * 2);
    _Float16* wd16   = (_Float16*)alloc((size_t)H_ * H_ * 2);
    float*    qkv    = (float*)   alloc((size_t)B_ * S_ * QKVW * 4);
    _Float16* qh     = (_Float16*)alloc((size_t)B_ * NH_ * S_ * HD_ * 2);
    _Float16* kh     = (_Float16*)alloc((size_t)B_ * G_ * S_ * HD_ * 2);
    _Float16* vt     = (_Float16*)alloc((size_t)B_ * G_ * S_ * HD_ * 2);
    _Float16* ctx    = (_Float16*)alloc((size_t)B_ * S_ * H_ * 2);

    auto cvt = [&](const float* i, _Float16* o, size_t n) {
        int n4 = (int)(n / 4);
        cvt_f32_f16<<<(n4 + 255) / 256, 256, 0, stream>>>(i, o, n4);
    };
    cvt(hs,     hs16,   (size_t)B_ * S_ * H_);
    cvt(wqkv,   wqkv16, (size_t)H_ * QKVW);
    cvt(wdense, wd16,   (size_t)H_ * H_);

    // QKV projection (+bias) -> fp32 scratch
    gemm_f16f32<<<dim3(QKVW / 128, (B_ * S_) / 128), 256, 0, stream>>>(
        hs16, wqkv16, bqkv, qkv, B_ * S_, QKVW, H_);

    // RoPE + layout split
    int n = B_ * S_ * QKVW;
    rope_split<<<(n + 255) / 256, 256, 0, stream>>>(qkv, rope, qh, kh, vt);

    // Causal GQA flash attention
    attn_kernel<<<dim3(S_ / 64, B_ * NH_), 128, 0, stream>>>(qh, kh, vt, ctx);

    // Output projection -> d_out fp32
    gemm_f16f32<<<dim3(H_ / 128, (B_ * S_) / 128), 256, 0, stream>>>(
        ctx, wd16, nullptr, out, B_ * S_, H_, H_);
}
